// LinearAttention_4544075399896
// MI455X (gfx1250) — matrix operand order repaired and checked by reading, not compiled, not run on hardware
//
#include <hip/hip_runtime.h>

typedef __attribute__((ext_vector_type(16))) __bf16 v16bf;
typedef __attribute__((ext_vector_type(8)))  __bf16 v8bf;
typedef __attribute__((ext_vector_type(8)))  float  v8f;
typedef __attribute__((__vector_size__(4 * sizeof(int)))) int v4i_b128;

static constexpr float SCALE = 0.125f; // 64^-0.5

#if __has_builtin(__builtin_amdgcn_global_load_async_to_lds_b128) && \
    __has_builtin(__builtin_amdgcn_s_wait_asynccnt)
#define HAS_ASYNC_LDS 1
#else
#define HAS_ASYNC_LDS 0
#endif

static __device__ __forceinline__ v16bf concat8(v8bf lo, v8bf hi) {
  return __builtin_shufflevector(lo, hi, 0, 1, 2, 3, 4, 5, 6, 7,
                                 8, 9, 10, 11, 12, 13, 14, 15);
}

// 16-byte global -> LDS copy; async (ASYNCcnt) when available
static __device__ __forceinline__ void lds_copy_b128(const __bf16* g, __bf16* l) {
#if HAS_ASYNC_LDS
  __builtin_amdgcn_global_load_async_to_lds_b128(
      (__attribute__((address_space(1))) v4i_b128*)(g),
      (__attribute__((address_space(3))) v4i_b128*)(l), 0, 0);
#else
  *(uint4*)l = *(const uint4*)g;
#endif
}
static __device__ __forceinline__ void lds_copy_join() {
#if HAS_ASYNC_LDS
  __builtin_amdgcn_s_wait_asynccnt(0);
#endif
}

// ---------------------------------------------------------------------------
// f32 -> bf16 convert (grid-stride)
// ---------------------------------------------------------------------------
__global__ void convert_f32_bf16(const float* __restrict__ in,
                                 __bf16* __restrict__ out, long long n) {
  long long i = (long long)blockIdx.x * blockDim.x + threadIdx.x;
  long long stride = (long long)gridDim.x * blockDim.x;
  for (; i < n; i += stride) out[i] = (__bf16)in[i];
}

// ---------------------------------------------------------------------------
// Generic tiled bf16 WMMA GEMM: C[M,N] = A[M,K] @ B[K,N] (+bias), batched by z
// Block = 256 threads (8 waves). Tile 128x128, K-stage 64 (2 WMMA K-steps).
// A staged via async global->LDS (double-buffered, one barrier per stage);
// B staged transposed through VGPRs ([n][k]) so both fragments are contiguous
// b128 LDS loads in contiguous k = 16h + s order (same order for A and B).
// ---------------------------------------------------------------------------
template <bool OUT_BF16, bool BIAS>
__global__ __launch_bounds__(256)
void gemm_bf16_kernel(const __bf16* __restrict__ A, const __bf16* __restrict__ B,
                      void* __restrict__ Cout, const float* __restrict__ bias,
                      int K, int lda, int ldb, int ldc,
                      long long strideA, long long strideB, long long strideC) {
  const int bx = blockIdx.x;   // N tile
  const int by = blockIdx.y;   // M tile
  const int bz = blockIdx.z;   // batch
  A += (long long)bz * strideA;
  B += (long long)bz * strideB;

  __shared__ __bf16 As[2][128][64];    // [m][k]  2 x 16 KB
  __shared__ __bf16 BsT[2][128][64];   // [n][k]  2 x 16 KB (transposed)

  const int tid  = threadIdx.x;
  const int lane = tid & 31;
  const int wave = tid >> 5;       // 0..7
  const int wm   = wave >> 1;      // 0..3 : 32 rows each
  const int wn   = wave & 1;       // 0..1 : 64 cols each

  v8f acc[2][4];
  for (int i = 0; i < 2; ++i)
    for (int j = 0; j < 4; ++j)
      for (int v = 0; v < 8; ++v) acc[i][j][v] = 0.0f;

  const int m_f  = lane & 15;          // fragment row (A) / col (B)
  const int khiA = (lane >> 4) << 3;   // +0 / +8   (unused)
  const int kb0B = (lane >> 4) << 4;   // +0 / +16  (B k offset)

  const __bf16* Abase = A + (long long)(by * 128) * lda;
  const __bf16* Bbase = B + (long long)bx * 128;

  v8bf bReg[4];
  auto load_B = [&](int k0) {
    const __bf16* Bg = Bbase + (long long)k0 * ldb;
#pragma unroll
    for (int i = 0; i < 4; ++i) {
      int s = tid + 256 * i;
      int rb = s >> 4, cb = (s & 15) * 8;        // B: k-row, n-col
      bReg[i] = *(const v8bf*)(Bg + (long long)rb * ldb + cb);
    }
  };
  auto stage_A = [&](int k0, int buf) {
    const __bf16* Ag = Abase + k0;
#pragma unroll
    for (int i = 0; i < 4; ++i) {
      int s = tid + 256 * i;
      int r = s >> 3, c = (s & 7) * 8;           // A: row, k-col
      lds_copy_b128(Ag + (long long)r * lda + c, &As[buf][r][c]);
    }
  };

  stage_A(0, 0);
  load_B(0);
  const int nStages = K >> 6;
  for (int i = 0; i < nStages; ++i) {
    const int cur = i & 1, nxt = cur ^ 1;
    lds_copy_join();                     // our As[cur] async parts landed
#pragma unroll
    for (int ii = 0; ii < 4; ++ii) {     // transpose-store B regs
      int s = tid + 256 * ii;
      int rb = s >> 4, cb = (s & 15) * 8;
#pragma unroll
      for (int e = 0; e < 8; ++e) BsT[cur][cb + e][rb] = bReg[ii][e];
    }
    __syncthreads();                     // As[cur]+BsT[cur] visible to all
    if (i + 1 < nStages) {               // next stage in flight during compute
      stage_A((i + 1) * 64, nxt);
      load_B((i + 1) * 64);
    }

#pragma unroll
    for (int kk = 0; kk < 2; ++kk) {
      v16bf af[2];
#pragma unroll
      for (int ti = 0; ti < 2; ++ti) {
        int row = wm * 32 + ti * 16 + m_f;
        af[ti] = *(const v16bf*)&As[cur][row][kk * 32 + kb0B];
      }
      v16bf bfr[4];
#pragma unroll
      for (int tj = 0; tj < 4; ++tj) {
        int col = wn * 64 + tj * 16 + m_f;
        bfr[tj] = *(const v16bf*)&BsT[cur][col][kk * 32 + kb0B];
      }
#pragma unroll
      for (int ti = 0; ti < 2; ++ti)
#pragma unroll
        for (int tj = 0; tj < 4; ++tj)
          acc[ti][tj] = __builtin_amdgcn_wmma_f32_16x16x32_bf16(
              false, af[ti], false, bfr[tj], (short)0, acc[ti][tj], false, false);
    }
  }

  // ---- epilogue (C/D layout: VGPR v -> M=v(+8 for hi lanes), N=lane&15) ----
  const int coff = lane & 15;
  const int roff = (lane >> 4) * 8;
  for (int ti = 0; ti < 2; ++ti)
    for (int tj = 0; tj < 4; ++tj) {
      int gr0 = by * 128 + wm * 32 + ti * 16 + roff;
      int gc  = bx * 128 + wn * 64 + tj * 16 + coff;
      float bval = BIAS ? bias[gc] : 0.0f;
#pragma unroll
      for (int v = 0; v < 8; ++v) {
        long long idx = (long long)bz * strideC + (long long)(gr0 + v) * ldc + gc;
        float val = acc[ti][tj][v] + bval;
        if (OUT_BF16) ((__bf16*)Cout)[idx] = (__bf16)val;
        else          ((float*)Cout)[idx]  = val;
      }
    }
}

// ---------------------------------------------------------------------------
// softmax over feature dim d=64 for q (in-place on bf16 qkv, cols 0..1023)
// ---------------------------------------------------------------------------
__global__ __launch_bounds__(256)
void softmax_q_kernel(__bf16* __restrict__ qkv) {
  long long gw = (long long)blockIdx.x * 8 + (threadIdx.x >> 5); // 0..524287
  long long rn = gw >> 4;          // 0..32767 (b*N + n)
  int h = (int)(gw & 15);
  int lane = threadIdx.x & 31;
  __bf16* p = qkv + rn * 3072 + h * 64;
  float v0 = (float)p[lane]      * SCALE;
  float v1 = (float)p[lane + 32] * SCALE;
  float m = fmaxf(v0, v1);
  for (int o = 16; o > 0; o >>= 1) m = fmaxf(m, __shfl_xor(m, o, 32));
  float e0 = __expf(v0 - m), e1 = __expf(v1 - m);
  float s = e0 + e1;
  for (int o = 16; o > 0; o >>= 1) s += __shfl_xor(s, o, 32);
  float inv = 1.0f / s;
  p[lane]      = (__bf16)(e0 * inv);
  p[lane + 32] = (__bf16)(e1 * inv);
}

// ---------------------------------------------------------------------------
// softmax over sequence dim N=8192 for k (in-place, cols 1024..2047)
// ---------------------------------------------------------------------------
__global__ __launch_bounds__(256)
void softmax_k_kernel(__bf16* __restrict__ qkv) {
  int col = blockIdx.x;                 // 0..4095
  int b = col >> 10, h = (col >> 6) & 15, e = col & 63;
  __bf16* p = qkv + ((long long)b * 8192) * 3072 + 1024 + h * 64 + e;
  int tid = threadIdx.x;
  __shared__ float red[256];

  float m = -3.402823466e38f;
  for (int n = tid; n < 8192; n += 256)
    m = fmaxf(m, (float)p[(long long)n * 3072]);
  red[tid] = m; __syncthreads();
  for (int s = 128; s > 0; s >>= 1) {
    if (tid < s) red[tid] = fmaxf(red[tid], red[tid + s]);
    __syncthreads();
  }
  m = red[0]; __syncthreads();

  float sum = 0.0f;
  for (int n = tid; n < 8192; n += 256)
    sum += __expf((float)p[(long long)n * 3072] - m);
  red[tid] = sum; __syncthreads();
  for (int s = 128; s > 0; s >>= 1) {
    if (tid < s) red[tid] += red[tid + s];
    __syncthreads();
  }
  float inv = 1.0f / red[0];

  for (int n = tid; n < 8192; n += 256) {
    long long off = (long long)n * 3072;
    p[off] = (__bf16)(__expf((float)p[off] - m) * inv);
  }
}

// ---------------------------------------------------------------------------
// partial context: ctxp[split][bh][d][e] = sum_{n in split} q'[n,d] * k'[n,e]
// grid (64 bh, 16 splits); 512 threads = 16 waves, one 16x16 WMMA tile each.
// Register-pipelined global loads; q/k tiles staged transposed for b128 frags.
// ---------------------------------------------------------------------------
__global__ __launch_bounds__(512)
void context_kernel(const __bf16* __restrict__ qkv, float* __restrict__ ctxp) {
  int bh = blockIdx.x;                  // 0..63
  int split = blockIdx.y;               // 0..15
  int b = bh >> 4, h = bh & 15;
  __shared__ __bf16 qsT[64][32];        // [d][n] 4 KB
  __shared__ __bf16 ksT[64][32];        // [e][n] 4 KB
  int tid = threadIdx.x;
  int lane = tid & 31;
  int wave = tid >> 5;                  // 0..15
  int ti = wave >> 2, tj = wave & 3;    // d-tile, e-tile

  v8f acc;
  for (int v = 0; v < 8; ++v) acc[v] = 0.0f;

  const __bf16* qbase = qkv + ((long long)b * 8192) * 3072 + h * 64; // q cols
  const __bf16* kbase = qbase + 1024;                                 // k cols
  const int m_f  = lane & 15;
  const int khiA = (lane >> 4) << 3;
  const int kb0B = (lane >> 4) << 4;

  // staging role: threads 0..255 -> q, 256..511 -> k; 256 segments of 8
  const __bf16* src = (tid < 256) ? qbase : kbase;
  __bf16(*dst)[32] = (tid < 256) ? qsT : ksT;
  const int seg = tid & 255;
  const int sr = seg >> 3;              // n row 0..31
  const int sc = (seg & 7) * 8;         // d/e col base

  const int nbase = split * 512;        // 16 iterations of 32 rows
  v8bf t = *(const v8bf*)(src + (long long)(nbase + sr) * 3072 + sc);
  for (int it = 0; it < 16; ++it) {
#pragma unroll
    for (int e = 0; e < 8; ++e) dst[sc + e][sr] = t[e];
    __syncthreads();
    v8bf tn = t;
    if (it + 1 < 16)                    // next tile in flight during compute
      tn = *(const v8bf*)(src + (long long)(nbase + (it + 1) * 32 + sr) * 3072 + sc);

    int row = ti * 16 + m_f;
    v16bf af = *(const v16bf*)&qsT[row][kb0B];
    v16bf bfr = *(const v16bf*)&ksT[tj * 16 + m_f][kb0B];

    acc = __builtin_amdgcn_wmma_f32_16x16x32_bf16(
        false, af, false, bfr, (short)0, acc, false, false);
    __syncthreads();
    t = tn;
  }

  int coff = lane & 15;
  int roff = (lane >> 4) * 8;
  float* out = ctxp + ((long long)split * 64 + bh) * 4096;
#pragma unroll
  for (int v = 0; v < 8; ++v)
    out[(ti * 16 + roff + v) * 64 + tj * 16 + coff] = acc[v];
}

// ---------------------------------------------------------------------------
// Mfold_b[h*64+d, o] = sum_e (sum_splits ctxp)[b,h][d,e] * w_out[h*64+e, o]
// 1 WG per (b,h); reduced C_h cached in LDS; W rows are L2-resident
// ---------------------------------------------------------------------------
__global__ __launch_bounds__(256)
void fold_kernel(const float* __restrict__ ctxp, const float* __restrict__ w_out,
                 __bf16* __restrict__ Mf) {
  int bh = blockIdx.x;                  // 0..63
  int b = bh >> 4, h = bh & 15;
  __shared__ float Cs[64][64];
  int tid = threadIdx.x;
  for (int s = tid; s < 4096; s += 256) {
    float a = 0.0f;
#pragma unroll
    for (int sp = 0; sp < 16; ++sp)
      a += ctxp[((long long)sp * 64 + bh) * 4096 + s];
    Cs[s >> 6][s & 63] = a;
  }
  __syncthreads();

  __bf16* out = Mf + ((long long)b * 1024 + h * 64) * 1024;
  const float* W = w_out + (long long)(h * 64) * 1024;
  for (int d = 0; d < 64; ++d)
    for (int j = 0; j < 4; ++j) {
      int o = tid + 256 * j;
      float a = 0.0f;
#pragma unroll 8
      for (int e = 0; e < 64; ++e) a += Cs[d][e] * W[(long long)e * 1024 + o];
      out[(long long)d * 1024 + o] = (__bf16)a;
    }
}

// ---------------------------------------------------------------------------
// Host launch
// ---------------------------------------------------------------------------
extern "C" void kernel_launch(void* const* d_in, const int* in_sizes, int n_in,
                              void* d_out, int out_size, void* d_ws, size_t ws_size,
                              hipStream_t stream) {
  (void)in_sizes; (void)n_in; (void)out_size; (void)ws_size;
  const float* x     = (const float*)d_in[0]; // [4,8192,1024]
  const float* w_qkv = (const float*)d_in[1]; // [1024,3072]
  const float* w_out = (const float*)d_in[2]; // [1024,1024]
  const float* b_out = (const float*)d_in[3]; // [1024]
  float* out = (float*)d_out;                 // [4,8192,1024]

  char* ws = (char*)d_ws;
  __bf16* x_bf  = (__bf16*)(ws);                    //  67,108,864 B
  __bf16* w_bf  = (__bf16*)(ws + 67108864);         //   6,291,456 B
  __bf16* qkv   = (__bf16*)(ws + 73400320);         // 201,326,592 B
  float*  ctxp  = (float*) (ws + 274726912);        //  16,777,216 B (16 splits)
  __bf16* Mf    = (__bf16*)(ws + 291504128);        //   8,388,608 B (total ~300 MB)

  // 1. convert inputs to bf16
  convert_f32_bf16<<<2048, 256, 0, stream>>>(x, x_bf, (long long)4 * 8192 * 1024);
  convert_f32_bf16<<<256, 256, 0, stream>>>(w_qkv, w_bf, (long long)1024 * 3072);

  // 2. qkv = x @ w_qkv  [32768 x 1024] @ [1024 x 3072] -> bf16
  gemm_bf16_kernel<true, false><<<dim3(3072 / 128, 32768 / 128, 1), 256, 0, stream>>>(
      x_bf, w_bf, (void*)qkv, nullptr, 1024, 1024, 3072, 3072, 0, 0, 0);

  // 3. softmax q over d (per row/head), 4. softmax k over N (per column)
  softmax_q_kernel<<<65536, 256, 0, stream>>>(qkv);
  softmax_k_kernel<<<4096, 256, 0, stream>>>(qkv);

  // 5. partial contexts (split-K over N, deterministic)
  context_kernel<<<dim3(64, 16), 512, 0, stream>>>(qkv, ctxp);

  // 6. reduce partials + fold context into w_out: Mf_b = blockdiag(C_b) @ W
  fold_kernel<<<64, 256, 0, stream>>>(ctxp, w_out, Mf);

  // 7. out_b = v_b @ Mf_b + b_out  (batched over b via grid.z)
  gemm_bf16_kernel<false, true><<<dim3(1024 / 128, 8192 / 128, 4), 256, 0, stream>>>(
      qkv + 2048, Mf, (void*)out, b_out, 1024, 3072, 1024, 1024,
      (long long)8192 * 3072, (long long)1024 * 1024, (long long)8192 * 1024);
}
